// UNetTFF_35476429865151
// MI455X (gfx1250) — compile-verified, matrix operand order checked by reading, not run on hardware
//
#include <hip/hip_runtime.h>

#define N_TOK 8192
#define NEXP  16
#define KTOP  4
#define BM    64
#define BN    128
#define BKK   32
#define CAP   (N_TOK * KTOP + NEXP * BM)   // 33792 padded assignment rows (fixed worst case)
#define GT    (CAP / BM)                   // 528 grouped M-tiles

typedef __attribute__((ext_vector_type(16))) __bf16 v16bf;
typedef __attribute__((ext_vector_type(8)))  float  v8f;

union Frag16 { v16bf v; uint4 q[2]; };
union Acc8   { v8f  v; float f[8]; };

__device__ __forceinline__ unsigned short f2bf(float f) {
    unsigned int u = __float_as_uint(f);
    u += 0x7fffu + ((u >> 16) & 1u);          // round-to-nearest-even f32 -> bf16
    return (unsigned short)(u >> 16);
}

__device__ __forceinline__ unsigned int pk2(float a, float b) {
    return (unsigned int)f2bf(a) | ((unsigned int)f2bf(b) << 16);
}

enum {
    FG_GROUPED = 1, FG_GATHER = 2, FG_SCATTER = 4, FG_ACC = 8,
    FG_DUAL = 16, FG_SCALE = 32, FG_BIAS = 64
};

// ---------------------------------------------------------------------------
// Generic tiled GEMM: C[M,N] = epilogue(A[M,K] @ B[K,N]).
//   DUAL    : second product acc2 = A @ B2; out = (acc+b1) * silu(acc2+b2)
//   GROUPED : M-space is per-expert padded segments (pad_off[17]); B += e*strideBe
//   GATHER  : A row taken from gidx[] (tokens gathered per expert; -1 = pad row,
//             address-clamped to row 0; its garbage output is never stored)
//   SCATTER : C row scatter-added to token gidx[row] (atomicAdd; -1 = skip)
//   SCALE   : multiply row by rowscale[row] (routing weight; 0 for pad rows)
// Hot path: uniform fullK branch -> unconditional clamped-address loads, no
// value selects, packed bf16 ds_store_b64; cold K-tail keeps zero-fill.
// ---------------------------------------------------------------------------
template <int F>
__global__ __launch_bounds__(256)
void gemm_k(const float* __restrict__ A, int lda,
            const float* __restrict__ B1, const float* __restrict__ B2, int ldb,
            long strideBe,
            const float* __restrict__ bias1, const float* __restrict__ bias2,
            float* __restrict__ C, int ldc,
            int M, int N, int K,
            const int* __restrict__ pad_off,
            const int* __restrict__ gidx,
            const float* __restrict__ rowscale)
{
    constexpr bool GROUPED = (F & FG_GROUPED) != 0;
    constexpr bool GATHER  = (F & FG_GATHER)  != 0;
    constexpr bool SCATTER = (F & FG_SCATTER) != 0;
    constexpr bool ACCUM   = (F & FG_ACC)     != 0;
    constexpr bool DUAL    = (F & FG_DUAL)    != 0;
    constexpr bool SCALE   = (F & FG_SCALE)   != 0;
    constexpr bool BIAS    = (F & FG_BIAS)    != 0;

    __shared__ unsigned short As [BM][BKK];   // 4 KB
    __shared__ unsigned short Bs1[BN][BKK];   // 8 KB (transposed: [n][k])
    __shared__ unsigned short Bs2[BN][BKK];   // 8 KB (only touched when DUAL)

    const int tid = threadIdx.x;
    const int r0  = blockIdx.x * BM;
    const int n0  = blockIdx.y * BN;

    int padTotal = M;
    const float* B1e = B1;
    const float* B2e = B2;
    if constexpr (GROUPED) {
        padTotal = pad_off[NEXP];
        if (r0 >= padTotal) return;
        int e = 0;
        while (e < NEXP - 1 && r0 >= pad_off[e + 1]) ++e;
        B1e = B1 + (size_t)e * (size_t)strideBe;
        if constexpr (DUAL) B2e = B2 + (size_t)e * (size_t)strideBe;
    }

    const int lane = tid & 31;
    const int r    = lane & 15;
    const int hi   = lane >> 4;
    const int wv   = tid >> 5;          // 8 waves
    const int wm   = (wv >> 2) * 32;    // wave M base (0/32)
    const int wn   = (wv & 3) * 32;     // wave N base (0/32/64/96)

    // B staging sub-block owned by this thread: 4 k-rows x 4 n-cols
    const int kb = (tid >> 5) * 4;      // 0..28
    const int nb = (tid & 31) * 4;      // 0..124

    // column address clamp (stays in-bounds; OOB columns never stored)
    int ncl[4];
#pragma unroll
    for (int jn = 0; jn < 4; ++jn) {
        int nn = n0 + nb + jn;
        ncl[jn] = (nn < N) ? nn : (N - 1);
    }

    Acc8 acc [2][2];
    Acc8 acc2[2][2];
#pragma unroll
    for (int mi = 0; mi < 2; ++mi)
#pragma unroll
        for (int ni = 0; ni < 2; ++ni)
#pragma unroll
            for (int j = 0; j < 8; ++j) {
                acc[mi][ni].f[j] = 0.f;
                if constexpr (DUAL) acc2[mi][ni].f[j] = 0.f;
            }

    for (int k0 = 0; k0 < K; k0 += BKK) {
        __syncthreads();
        const bool fullK = (k0 + BKK) <= K;     // wave-uniform (scalar branch)

        // ---- stage A (f32 -> bf16), 64x32, packed b64 stores ----
#pragma unroll
        for (int it = 0; it < 2; ++it) {
            int q   = tid + it * 256;
            int row = q >> 3;
            int kq  = (q & 7) * 4;
            int grow = r0 + row;
            int arow = grow;
            if constexpr (GATHER) {
                int gi = gidx[grow];            // always in-bounds (grow < CAP)
                arow = (gi >= 0) ? gi : 0;      // address clamp, no value select
            }
            const float* rowp = A + (size_t)arow * (size_t)lda;
            float e0, e1, e2, e3;
            if (fullK) {
                const float* p = rowp + k0 + kq;
                e0 = p[0]; e1 = p[1]; e2 = p[2]; e3 = p[3];
            } else {
                int kk0 = k0 + kq;
                int c0 = (kk0     < K) ? kk0     : K - 1;
                int c1 = (kk0 + 1 < K) ? kk0 + 1 : K - 1;
                int c2 = (kk0 + 2 < K) ? kk0 + 2 : K - 1;
                int c3 = (kk0 + 3 < K) ? kk0 + 3 : K - 1;
                float t0 = rowp[c0], t1 = rowp[c1], t2 = rowp[c2], t3 = rowp[c3];
                e0 = (kk0     < K) ? t0 : 0.f;
                e1 = (kk0 + 1 < K) ? t1 : 0.f;
                e2 = (kk0 + 2 < K) ? t2 : 0.f;
                e3 = (kk0 + 3 < K) ? t3 : 0.f;
            }
            uint2 pk; pk.x = pk2(e0, e1); pk.y = pk2(e2, e3);
            *(uint2*)&As[row][kq] = pk;
        }

        // ---- stage B (f32 -> bf16, register-transposed 4x4 sub-blocks) ----
        {
            if (k0 + BKK < K)
                __builtin_prefetch(B1e + (size_t)(k0 + BKK) * (size_t)ldb + ncl[0], 0, 1);
            float v[4][4];
            if (fullK) {
#pragma unroll
                for (int jk = 0; jk < 4; ++jk) {
                    const float* bp = B1e + (size_t)(k0 + kb + jk) * (size_t)ldb;
#pragma unroll
                    for (int jn = 0; jn < 4; ++jn) v[jk][jn] = bp[ncl[jn]];
                }
            } else {
#pragma unroll
                for (int jk = 0; jk < 4; ++jk) {
                    int kk = k0 + kb + jk;
                    int kc = (kk < K) ? kk : (K - 1);
                    const float* bp = B1e + (size_t)kc * (size_t)ldb;
#pragma unroll
                    for (int jn = 0; jn < 4; ++jn) {
                        float t = bp[ncl[jn]];
                        v[jk][jn] = (kk < K) ? t : 0.f;
                    }
                }
            }
#pragma unroll
            for (int jn = 0; jn < 4; ++jn) {
                uint2 pk;
                pk.x = pk2(v[0][jn], v[1][jn]);
                pk.y = pk2(v[2][jn], v[3][jn]);
                *(uint2*)&Bs1[nb + jn][kb] = pk;
            }
            if constexpr (DUAL) {
                float w[4][4];
                if (fullK) {
#pragma unroll
                    for (int jk = 0; jk < 4; ++jk) {
                        const float* bp2 = B2e + (size_t)(k0 + kb + jk) * (size_t)ldb;
#pragma unroll
                        for (int jn = 0; jn < 4; ++jn) w[jk][jn] = bp2[ncl[jn]];
                    }
                } else {
#pragma unroll
                    for (int jk = 0; jk < 4; ++jk) {
                        int kk = k0 + kb + jk;
                        int kc = (kk < K) ? kk : (K - 1);
                        const float* bp2 = B2e + (size_t)kc * (size_t)ldb;
#pragma unroll
                        for (int jn = 0; jn < 4; ++jn) {
                            float t = bp2[ncl[jn]];
                            w[jk][jn] = (kk < K) ? t : 0.f;
                        }
                    }
                }
#pragma unroll
                for (int jn = 0; jn < 4; ++jn) {
                    uint2 pk;
                    pk.x = pk2(w[0][jn], w[1][jn]);
                    pk.y = pk2(w[2][jn], w[3][jn]);
                    *(uint2*)&Bs2[nb + jn][kb] = pk;
                }
            }
        }
        __syncthreads();

        // ---- fragments: two contiguous b128 LDS loads per operand ----
        Frag16 a[2], b[2], b2[2];
#pragma unroll
        for (int mi = 0; mi < 2; ++mi) {
            int rowL = wm + mi * 16 + r;
            a[mi].q[0] = *(const uint4*)&As[rowL][hi * 8];
            a[mi].q[1] = *(const uint4*)&As[rowL][hi * 8 + 16];
        }
#pragma unroll
        for (int ni = 0; ni < 2; ++ni) {
            int colL = wn + ni * 16 + r;
            b[ni].q[0] = *(const uint4*)&Bs1[colL][hi * 8];
            b[ni].q[1] = *(const uint4*)&Bs1[colL][hi * 8 + 16];
            if constexpr (DUAL) {
                b2[ni].q[0] = *(const uint4*)&Bs2[colL][hi * 8];
                b2[ni].q[1] = *(const uint4*)&Bs2[colL][hi * 8 + 16];
            }
        }
#pragma unroll
        for (int mi = 0; mi < 2; ++mi)
#pragma unroll
            for (int ni = 0; ni < 2; ++ni) {
                acc[mi][ni].v = __builtin_amdgcn_wmma_f32_16x16x32_bf16(
                    false, a[mi].v, false, b[ni].v, (short)0, acc[mi][ni].v, false, false);
                if constexpr (DUAL)
                    acc2[mi][ni].v = __builtin_amdgcn_wmma_f32_16x16x32_bf16(
                        false, a[mi].v, false, b2[ni].v, (short)0, acc2[mi][ni].v, false, false);
            }
    }

    // ---- epilogue ----
#pragma unroll
    for (int mi = 0; mi < 2; ++mi)
#pragma unroll
        for (int ni = 0; ni < 2; ++ni)
#pragma unroll
            for (int j = 0; j < 8; ++j) {
                int grow = r0 + wm + mi * 16 + hi * 8 + j;
                int gcol = n0 + wn + ni * 16 + r;
                bool rowok = GROUPED ? (grow < padTotal) : (grow < M);
                if (!rowok || gcol >= N) continue;
                float v1 = acc[mi][ni].f[j];
                if constexpr (BIAS) v1 += bias1[gcol];
                float outv;
                if constexpr (DUAL) {
                    float v2 = acc2[mi][ni].f[j];
                    if constexpr (BIAS) v2 += bias2[gcol];
                    float sg = 1.0f / (1.0f + __expf(-v2));
                    outv = v1 * v2 * sg;                  // v1 * silu(v2)
                    if constexpr (SCALE) outv *= rowscale[grow];
                } else {
                    outv = v1;
                }
                if constexpr (SCATTER) {
                    int t = gidx[grow];
                    if (t >= 0) atomicAdd(&C[(size_t)t * (size_t)ldc + gcol], outv);
                } else if constexpr (ACCUM) {
                    C[(size_t)grow * (size_t)ldc + gcol] += outv;
                } else {
                    C[(size_t)grow * (size_t)ldc + gcol] = outv;
                }
            }
}

// ---------------------------------------------------------------------------
// MoE routing: softmax over 16 logits, greedy top-4, normalize, count experts.
// ---------------------------------------------------------------------------
__global__ void router_k(const float* __restrict__ x, int d,
                         const float* __restrict__ gate,
                         int* __restrict__ topi, float* __restrict__ topv,
                         int* __restrict__ cnt, int n)
{
    int t = blockIdx.x * blockDim.x + threadIdx.x;
    if (t >= n) return;
    const float* xr = x + (size_t)t * d;
    float l[NEXP];
    for (int e = 0; e < NEXP; ++e) {
        const float* g = gate + (size_t)e * d;
        float s = 0.f;
        for (int k = 0; k < d; ++k) s += xr[k] * g[k];
        l[e] = s;
    }
    float mx = l[0];
    for (int e = 1; e < NEXP; ++e) mx = fmaxf(mx, l[e]);
    float sum = 0.f;
    for (int e = 0; e < NEXP; ++e) { l[e] = __expf(l[e] - mx); sum += l[e]; }
    float inv = 1.f / sum;
    for (int e = 0; e < NEXP; ++e) l[e] *= inv;

    int ei[KTOP]; float ev[KTOP]; float s4 = 0.f;
    for (int k = 0; k < KTOP; ++k) {
        int best = 0; float bv = -1.f;
        for (int e = 0; e < NEXP; ++e)
            if (l[e] > bv) { bv = l[e]; best = e; }
        l[best] = -2.f;
        ei[k] = best; ev[k] = bv; s4 += bv;
    }
    float invs = 1.0f / s4;   // ROUTED_SCALE = 1.0
    for (int k = 0; k < KTOP; ++k) {
        topi[t * KTOP + k] = ei[k];
        topv[t * KTOP + k] = ev[k] * invs;
        atomicAdd(&cnt[ei[k]], 1);
    }
}

__global__ void offsets_k(const int* __restrict__ cnt, int* __restrict__ pad_off,
                          int* __restrict__ cur)
{
    if (threadIdx.x == 0 && blockIdx.x == 0) {
        int o = 0;
        for (int e = 0; e < NEXP; ++e) {
            pad_off[e] = o;
            o += ((cnt[e] + BM - 1) / BM) * BM;
            cur[e] = 0;
        }
        pad_off[NEXP] = o;
    }
}

__global__ void scatter_k(const int* __restrict__ topi, const float* __restrict__ topv,
                          const int* __restrict__ pad_off, int* __restrict__ cur,
                          int* __restrict__ aidx, float* __restrict__ aw, int n)
{
    int i = blockIdx.x * blockDim.x + threadIdx.x;
    if (i >= n * KTOP) return;
    int e = topi[i];
    int p = atomicAdd(&cur[e], 1);
    int pos = pad_off[e] + p;
    aidx[pos] = i / KTOP;
    aw[pos] = topv[i];
}

__global__ void zero_f32_k(float* __restrict__ p, long n)
{
    long i = blockIdx.x * (long)blockDim.x + threadIdx.x;
    if (i < n) p[i] = 0.f;
}

__global__ void fill_i32_k(int* __restrict__ p, int v, int n)
{
    int i = blockIdx.x * blockDim.x + threadIdx.x;
    if (i < n) p[i] = v;
}

__global__ void concat_k(const float* __restrict__ a, int da,
                         const float* __restrict__ b, int db,
                         float* __restrict__ o, int n)
{
    int dt = da + db;
    long i = blockIdx.x * (long)blockDim.x + threadIdx.x;
    if (i >= (long)n * dt) return;
    int row = (int)(i / dt), c = (int)(i % dt);
    o[i] = (c < da) ? a[(size_t)row * da + c] : b[(size_t)row * db + (c - da)];
}

// ---------------------------------------------------------------------------
// Host orchestration
// ---------------------------------------------------------------------------
static const int encd_[8] = {768, 662, 556, 450, 344, 238, 132, 27};
static const int decd_[8] = {27, 132, 238, 344, 450, 556, 662, 768};

struct LinP { const float* w; const float* b; };
struct BlockP {
    LinP lin0, sw0, tl0, tsw0, tl1, tsw1;
    const float *gate, *w1, *w3, *w2, *sw1, *sw3, *sw2;
};

struct Cur { void* const* din; const int* sz; int n; int i; bool ok; bool strict; };

static const float* takeF(Cur& c, long expect)
{
    if (c.i >= c.n) { c.ok = false; return nullptr; }
    if (c.strict && (long)c.sz[c.i] != expect) c.ok = false;
    return (const float*)c.din[c.i++];
}

static void blockDims(int idx, int& din, int& dout)
{
    if (idx < 7)       { din = encd_[idx];        dout = encd_[idx + 1]; }
    else if (idx == 7) { din = 27;                dout = 27; }
    else               { din = 2 * decd_[idx - 8]; dout = decd_[idx - 7]; }
}

// dict-insertion order: lin0(w,b), sw0(w,b), moe(gate,w1,w3,w2,sw1,sw3,sw2),
// tail_lin[0..1](w,b), tail_sw[0..1](w,b)
static void parse_block_A(Cur& c, BlockP& b, int din, int d)
{
    long dd = (long)d * d;
    b.lin0.w = takeF(c, (long)din * d); b.lin0.b = takeF(c, d);
    b.sw0.w  = takeF(c, 2 * dd);        b.sw0.b  = takeF(c, 2L * d);
    b.gate = takeF(c, (long)NEXP * d);
    b.w1 = takeF(c, NEXP * dd); b.w3 = takeF(c, NEXP * dd); b.w2 = takeF(c, NEXP * dd);
    b.sw1 = takeF(c, 2 * dd); b.sw3 = takeF(c, 2 * dd); b.sw2 = takeF(c, 2 * dd);
    b.tl0.w = takeF(c, dd); b.tl0.b = takeF(c, d);
    b.tl1.w = takeF(c, dd); b.tl1.b = takeF(c, d);
    b.tsw0.w = takeF(c, 2 * dd); b.tsw0.b = takeF(c, 2L * d);
    b.tsw1.w = takeF(c, 2 * dd); b.tsw1.b = takeF(c, 2L * d);
}

// jax sorted-key order: lin0(b,w), moe(gate,sw1,sw2,sw3,w1,w2,w3), sw0(b,w),
// tail_lin[0..1](b,w), tail_sw[0..1](b,w)
static void parse_block_B(Cur& c, BlockP& b, int din, int d)
{
    long dd = (long)d * d;
    b.lin0.b = takeF(c, d); b.lin0.w = takeF(c, (long)din * d);
    b.gate = takeF(c, (long)NEXP * d);
    b.sw1 = takeF(c, 2 * dd); b.sw2 = takeF(c, 2 * dd); b.sw3 = takeF(c, 2 * dd);
    b.w1 = takeF(c, NEXP * dd); b.w2 = takeF(c, NEXP * dd); b.w3 = takeF(c, NEXP * dd);
    b.sw0.b = takeF(c, 2L * d); b.sw0.w = takeF(c, 2 * dd);
    b.tl0.b = takeF(c, d); b.tl0.w = takeF(c, dd);
    b.tl1.b = takeF(c, d); b.tl1.w = takeF(c, dd);
    b.tsw0.b = takeF(c, 2L * d); b.tsw0.w = takeF(c, 2 * dd);
    b.tsw1.b = takeF(c, 2L * d); b.tsw1.w = takeF(c, 2 * dd);
}

static bool parseAll_A(void* const* din, const int* sz, int n, BlockP* blk,
                       const float*& x, bool strict)
{
    Cur c{din, sz, n, 0, true, strict};
    x = takeF(c, (long)N_TOK * 768);
    for (int i = 0; i < 15; ++i) { int a, b; blockDims(i, a, b); parse_block_A(c, blk[i], a, b); }
    return c.ok && c.i == n;
}

static bool parseAll_B(void* const* din, const int* sz, int n, BlockP* blk,
                       const float*& x, bool strict)
{
    Cur c{din, sz, n, 0, true, strict};
    int a, b;
    blockDims(7, a, b); parse_block_B(c, blk[7], a, b);                   // bottleneck
    for (int i = 8; i < 15; ++i) { blockDims(i, a, b); parse_block_B(c, blk[i], a, b); } // decoder
    for (int i = 0; i < 7; ++i)  { blockDims(i, a, b); parse_block_B(c, blk[i], a, b); } // encoder
    x = takeF(c, (long)N_TOK * 768);
    return c.ok && c.i == n;
}

struct Scr {
    float *t1, *t2, *t3, *hbuf, *topv, *aw;
    int *topi, *cnt, *cur, *padoff, *aidx;
    hipStream_t s;
};

static inline int ceildiv(int a, int b) { return (a + b - 1) / b; }

static void run_block(const BlockP& P, int din, int d,
                      const float* xin, float* xout, const Scr& S)
{
    hipStream_t s = S.s;
    const int M = N_TOK;
    dim3 T(256);

    // 1) t1 = xin @ lin0.w + b
    gemm_k<FG_BIAS><<<dim3(M / BM, ceildiv(d, BN)), T, 0, s>>>(
        xin, din, P.lin0.w, nullptr, d, 0, P.lin0.b, nullptr,
        S.t1, d, M, d, din, nullptr, nullptr, nullptr);
    // 2) t2 = swiglu(t1 @ sw0.w + b) : a * silu(b)
    gemm_k<FG_DUAL | FG_BIAS><<<dim3(M / BM, ceildiv(d, BN)), T, 0, s>>>(
        S.t1, d, P.sw0.w, P.sw0.w + d, 2 * d, 0, P.sw0.b, P.sw0.b + d,
        S.t2, d, M, d, d, nullptr, nullptr, nullptr);

    // 3) MoE on t2 -> t3
    zero_f32_k<<<ceildiv(M * d, 256), T, 0, s>>>(S.t3, (long)M * d);
    fill_i32_k<<<1, 64, 0, s>>>(S.cnt, 0, NEXP);
    router_k<<<ceildiv(M, 256), T, 0, s>>>(S.t2, d, P.gate, S.topi, S.topv, S.cnt, M);
    offsets_k<<<1, 1, 0, s>>>(S.cnt, S.padoff, S.cur);
    fill_i32_k<<<ceildiv(CAP, 256), T, 0, s>>>(S.aidx, -1, CAP);
    zero_f32_k<<<ceildiv(CAP, 256), T, 0, s>>>(S.aw, (long)CAP);
    scatter_k<<<ceildiv(M * KTOP, 256), T, 0, s>>>(S.topi, S.topv, S.padoff, S.cur,
                                                   S.aidx, S.aw, M);
    // routed hidden g = silu(x@w1) * (x@w3) * routing_weight  (grouped, gathered A)
    gemm_k<FG_GROUPED | FG_GATHER | FG_DUAL | FG_SCALE>
        <<<dim3(GT, ceildiv(d, BN)), T, 0, s>>>(
        S.t2, d, P.w3, P.w1, d, (long)d * d, nullptr, nullptr,
        S.hbuf, d, CAP, d, d, S.padoff, S.aidx, S.aw);
    // routed out: t3[token] += g @ w2[e]  (grouped, atomic scatter)
    gemm_k<FG_GROUPED | FG_SCATTER><<<dim3(GT, ceildiv(d, BN)), T, 0, s>>>(
        S.hbuf, d, P.w2, nullptr, d, (long)d * d, nullptr, nullptr,
        S.t3, d, CAP, d, d, S.padoff, S.aidx, nullptr);
    // shared experts: t1 = silu(x@sw1) * (x@sw3)   [M, 2d]
    gemm_k<FG_DUAL><<<dim3(M / BM, ceildiv(2 * d, BN)), T, 0, s>>>(
        S.t2, d, P.sw3, P.sw1, 2 * d, 0, nullptr, nullptr,
        S.t1, 2 * d, M, 2 * d, d, nullptr, nullptr, nullptr);
    // t3 += t1 @ sw2
    gemm_k<FG_ACC><<<dim3(M / BM, ceildiv(d, BN)), T, 0, s>>>(
        S.t1, 2 * d, P.sw2, nullptr, d, 0, nullptr, nullptr,
        S.t3, d, M, d, 2 * d, nullptr, nullptr, nullptr);

    // 4) two tail lin+SwiGLU stages
    gemm_k<FG_BIAS><<<dim3(M / BM, ceildiv(d, BN)), T, 0, s>>>(
        S.t3, d, P.tl0.w, nullptr, d, 0, P.tl0.b, nullptr,
        S.t1, d, M, d, d, nullptr, nullptr, nullptr);
    gemm_k<FG_DUAL | FG_BIAS><<<dim3(M / BM, ceildiv(d, BN)), T, 0, s>>>(
        S.t1, d, P.tsw0.w, P.tsw0.w + d, 2 * d, 0, P.tsw0.b, P.tsw0.b + d,
        S.t2, d, M, d, d, nullptr, nullptr, nullptr);
    gemm_k<FG_BIAS><<<dim3(M / BM, ceildiv(d, BN)), T, 0, s>>>(
        S.t2, d, P.tl1.w, nullptr, d, 0, P.tl1.b, nullptr,
        S.t1, d, M, d, d, nullptr, nullptr, nullptr);
    gemm_k<FG_DUAL | FG_BIAS><<<dim3(M / BM, ceildiv(d, BN)), T, 0, s>>>(
        S.t1, d, P.tsw1.w, P.tsw1.w + d, 2 * d, 0, P.tsw1.b, P.tsw1.b + d,
        xout, d, M, d, d, nullptr, nullptr, nullptr);
}

extern "C" void kernel_launch(void* const* d_in, const int* in_sizes, int n_in,
                              void* d_out, int out_size, void* d_ws, size_t ws_size,
                              hipStream_t stream)
{
    (void)out_size; (void)ws_size;

    BlockP blk[15];
    const float* x = nullptr;
    if (!parseAll_A(d_in, in_sizes, n_in, blk, x, true))
        if (!parseAll_B(d_in, in_sizes, n_in, blk, x, true))
            parseAll_A(d_in, in_sizes, n_in, blk, x, false);   // best effort

    // ---- carve workspace ----
    char* base = (char*)d_ws;
    size_t off = 0;
    auto alloc = [&](size_t bytes) -> void* {
        off = (off + 255) & ~(size_t)255;
        void* p = base + off;
        off += bytes;
        return p;
    };
    float* skip[7];
    for (int i = 0; i < 7; ++i)
        skip[i] = (float*)alloc((size_t)N_TOK * encd_[i + 1] * 4);
    float* xa   = (float*)alloc((size_t)N_TOK * 768 * 4);
    float* cbuf = (float*)alloc((size_t)N_TOK * 1536 * 4);

    Scr S;
    S.t1   = (float*)alloc((size_t)N_TOK * 1536 * 4);
    S.t2   = (float*)alloc((size_t)N_TOK * 768 * 4);
    S.t3   = (float*)alloc((size_t)N_TOK * 768 * 4);
    S.hbuf = (float*)alloc((size_t)CAP * 768 * 4);
    S.topi = (int*)alloc((size_t)N_TOK * KTOP * 4);
    S.topv = (float*)alloc((size_t)N_TOK * KTOP * 4);
    S.cnt    = (int*)alloc(64);
    S.cur    = (int*)alloc(64);
    S.padoff = (int*)alloc(68);
    S.aidx = (int*)alloc((size_t)CAP * 4);
    S.aw   = (float*)alloc((size_t)CAP * 4);
    S.s = stream;

    // ---- encoder ----
    const float* curx = x;
    for (int i = 0; i < 7; ++i) {
        run_block(blk[i], encd_[i], encd_[i + 1], curx, skip[i], S);
        curx = skip[i];
    }
    // ---- bottleneck ----
    run_block(blk[7], 27, 27, skip[6], xa, S);
    // ---- decoder (with skip concat) ----
    float* cx = xa;
    for (int i = 0; i < 7; ++i) {
        int da = decd_[i];
        concat_k<<<ceildiv(N_TOK * 2 * da, 256), 256, 0, stream>>>(
            cx, da, skip[6 - i], da, cbuf, N_TOK);
        float* outp = (i == 6) ? (float*)d_out : xa;
        run_block(blk[8 + i], 2 * da, decd_[i + 1], cbuf, outp, S);
        cx = outp;
    }
}
